// Llama4TextAttention_9483287789700
// MI455X (gfx1250) — matrix operand order repaired and checked by reading, not compiled, not run on hardware
//
#include <hip/hip_runtime.h>
#include <hip/hip_bf16.h>

// MI455X / gfx1250, wave32, WMMA bf16 path + TDM staging.
typedef __bf16 bf16;
typedef __attribute__((ext_vector_type(16))) __bf16    v16bf;
typedef __attribute__((ext_vector_type(8)))  float     v8f;
typedef __attribute__((ext_vector_type(4)))  unsigned  v4u;
typedef __attribute__((ext_vector_type(8)))  int       v8i;
typedef __attribute__((ext_vector_type(4)))  int       v4i;

#define WMMA_BF16(A, Bm, C) \
    __builtin_amdgcn_wmma_f32_16x16x32_bf16(false, (A), false, (Bm), (short)0, (C), false, false)

#if defined(__has_builtin)
#  if __has_builtin(__builtin_amdgcn_tensor_load_to_lds)
#    define HAVE_TDM 1
#  endif
#endif

#ifdef HAVE_TDM
// ---------------------------------------------------------------------------
// TDM 2D tile load: rows x 128 bf16, contiguous rows in global memory,
// LDS destination rows padded to 136 bf16 (row = 64 DWORDs -> pad_interval=5,
// pad 4 DWORDs -> pad_amount=3). Tracked with TENSORcnt.
// clang-23 / therock-10.0 builtin arity: (v4u, v8i, v4i, v4i, v8i, i32 cpol).
// ---------------------------------------------------------------------------
__device__ __forceinline__ void tdm_load_2d_bf16(unsigned lds_addr, const bf16* gaddr,
                                                 int rows)
{
    const unsigned long long ga = (unsigned long long)(const void*)gaddr;
    v4u g0;
    g0[0] = 1u;                                            // count=1 (valid user D#)
    g0[1] = lds_addr;                                      // LDS byte address
    g0[2] = (unsigned)(ga & 0xFFFFFFFFull);                // global_addr[31:0]
    g0[3] = (unsigned)((ga >> 32) & 0x1FFFFFFull)          // global_addr[56:32]
          | (2u << 30);                                    // type=2 ("image")
    v8i g1;
    g1[0] = (int)((1u << 16)                               // data_size = 2 bytes
                | (1u << 20)                               // pad_enable
                | (5u << 22)                               // pad_interval: 64 DWORDs
                | (3u << 25));                             // pad_amount: 4 DWORDs
    g1[1] = (int)((128u & 0xFFFFu) << 16);                 // tensor_dim0 lo16 @bits63:48
    g1[2] = (int)((128u >> 16) | ((2048u & 0xFFFFu) << 16)); // dim0 hi | tensor_dim1 lo
    g1[3] = (int)((2048u >> 16) | (128u << 16));           // dim1 hi | tile_dim0 = 128
    g1[4] = (int)(rows & 0xFFFF);                          // tile_dim1 = rows, tile_dim2=0
    g1[5] = 128;                                           // tensor_dim0_stride lo32
    g1[6] = 0;                                             // stride0 hi | stride1 lo
    g1[7] = 0;                                             // stride1 hi
    const v4i z4 = {0, 0, 0, 0};
    const v8i z8 = {0, 0, 0, 0, 0, 0, 0, 0};
    __builtin_amdgcn_tensor_load_to_lds(g0, g1, z4, z4, z8, 0);
}
#endif

// ---------------------------------------------------------------------------
// Tiled GEMM: C(f32, MxN) = A(AT, MxK, row-major) * W(f32, KxN, row-major)
// Block tile 128x128, K-step 32, 8 waves, each wave 64x32 (4x2 WMMA tiles).
// fp32 -> bf16 conversion happens on the global->LDS staging path.
// ---------------------------------------------------------------------------
template <typename AT>
__global__ void __launch_bounds__(256)
gemm_bf16_wmma(const AT* __restrict__ A, const float* __restrict__ W,
               float* __restrict__ C, int M, int N, int K)
{
    constexpr int LDA = 40;    // padded bf16 stride for A tile
    constexpr int LDB = 136;   // padded bf16 stride for B tile
    __shared__ bf16 As[128 * LDA];
    __shared__ bf16 Bs[32 * LDB];

    const int tid  = threadIdx.x;
    const int lane = tid & 31;
    const int wave = tid >> 5;
    const int wm   = (wave >> 2) * 64;   // wave M offset: 0 / 64
    const int wn   = (wave & 3) * 32;    // wave N offset: 0 / 32 / 64 / 96
    const int m0   = blockIdx.y * 128;
    const int n0   = blockIdx.x * 128;
    const int lm   = lane & 15;
    const int hi   = lane >> 4;

    v8f acc[4][2];
    #pragma unroll
    for (int i = 0; i < 4; i++)
        #pragma unroll
        for (int j = 0; j < 2; j++)
            #pragma unroll
            for (int e = 0; e < 8; e++) acc[i][j][e] = 0.0f;

    for (int k0 = 0; k0 < K; k0 += 32) {
        // ---- prefetch next K-tile (weights resident in 192MB L2) ----
        if (k0 + 32 < K) {
            __builtin_prefetch(A + (size_t)(m0 + (tid >> 1)) * K + (k0 + 32) + (tid & 1) * 16, 0, 1);
            __builtin_prefetch(W + (size_t)(k0 + 32 + (tid >> 5)) * N + n0 + (tid & 31) * 4, 0, 1);
        }
        // ---- stage A tile (128 x 32) into LDS as bf16 ----
        if constexpr (sizeof(AT) == 4) {
            #pragma unroll
            for (int p = 0; p < 4; p++) {
                int idx = tid + p * 256;             // 1024 float4 chunks
                int r = idx >> 3, c4 = idx & 7;
                const float4 v = *(const float4*)(A + (size_t)(m0 + r) * K + k0 + c4 * 4);
                bf16* d = &As[r * LDA + c4 * 4];
                d[0] = (bf16)v.x; d[1] = (bf16)v.y; d[2] = (bf16)v.z; d[3] = (bf16)v.w;
            }
        } else {
            #pragma unroll
            for (int p = 0; p < 4; p++) {
                int idx = tid + p * 256;             // 1024 x (4 bf16) chunks
                int r = idx >> 3, c = idx & 7;
                *(ushort4*)&As[r * LDA + c * 4] =
                    *(const ushort4*)(A + (size_t)(m0 + r) * K + k0 + c * 4);
            }
        }
        // ---- stage B tile (32 x 128) into LDS as bf16 ----
        #pragma unroll
        for (int p = 0; p < 4; p++) {
            int idx = tid + p * 256;                 // 1024 float4 chunks
            int r = idx >> 5, c4 = idx & 31;
            const float4 v = *(const float4*)(W + (size_t)(k0 + r) * N + n0 + c4 * 4);
            bf16* d = &Bs[r * LDB + c4 * 4];
            d[0] = (bf16)v.x; d[1] = (bf16)v.y; d[2] = (bf16)v.z; d[3] = (bf16)v.w;
        }
        __syncthreads();

        // ---- A fragments (16x32 bf16 per-lane layout) ----
        v16bf af[4];
        #pragma unroll
        for (int mi = 0; mi < 4; mi++) {
            const int m = wm + mi * 16 + lm;
            #pragma unroll
            for (int v = 0; v < 8; v++) {
                const int kb = ((v >> 2) << 4) + (hi << 3) + ((v & 3) << 1);
                af[mi][2 * v]     = As[m * LDA + kb];
                af[mi][2 * v + 1] = As[m * LDA + kb + 1];
            }
        }
        #pragma unroll
        for (int ni = 0; ni < 2; ni++) {
            v16bf bfr;
            const int n = wn + ni * 16 + lm;
            #pragma unroll
            for (int v = 0; v < 8; v++) {
                const int k = ((v >> 2) << 4) + (hi << 3) + ((v & 3) << 1);
                bfr[2 * v]     = Bs[k * LDB + n];
                bfr[2 * v + 1] = Bs[(k + 1) * LDB + n];
            }
            #pragma unroll
            for (int mi = 0; mi < 4; mi++)
                acc[mi][ni] = WMMA_BF16(af[mi], bfr, acc[mi][ni]);
        }
        __syncthreads();
    }

    // ---- store C (f32, per 16x16 C/D layout) ----
    #pragma unroll
    for (int mi = 0; mi < 4; mi++)
        #pragma unroll
        for (int ni = 0; ni < 2; ni++) {
            const int n = n0 + wn + ni * 16 + lm;
            #pragma unroll
            for (int j = 0; j < 8; j++) {
                const int m = m0 + wm + mi * 16 + hi * 8 + j;
                C[(size_t)m * N + n] = acc[mi][ni][j];
            }
        }
}

// ---------------------------------------------------------------------------
// RoPE (interleaved pairs) + L2 head-norm + transpose to head-major bf16.
// One block per (b,l). Wave handles one head's 128-dim vector (64 pairs,
// 2 pairs/lane); sum-of-squares reduced across the wave32.
// ---------------------------------------------------------------------------
__global__ void __launch_bounds__(256)
rope_norm_transpose(const float* __restrict__ q_raw, const float* __restrict__ k_raw,
                    const float* __restrict__ v_raw, const float* __restrict__ fcos,
                    const float* __restrict__ fsin,
                    bf16* __restrict__ q_t, bf16* __restrict__ k_t, bf16* __restrict__ v_t)
{
    const int bl   = blockIdx.x;          // b*2048 + l
    const int b    = bl >> 11;
    const int l    = bl & 2047;
    const int tid  = threadIdx.x;
    const int wave = tid >> 5;
    const int lane = tid & 31;

    const int p0 = lane * 2, p1 = lane * 2 + 1;
    const float c0 = fcos[l * 64 + p0], s0 = fsin[l * 64 + p0];
    const float c1 = fcos[l * 64 + p1], s1 = fsin[l * 64 + p1];

    // ---- Q: 32 heads, each wave does heads {wave, wave+8, wave+16, wave+24}
    #pragma unroll
    for (int hh = 0; hh < 4; hh++) {
        const int h = wave + hh * 8;
        const float* src = q_raw + (size_t)bl * 4096 + h * 128;
        const float xr0 = src[2 * p0], xi0 = src[2 * p0 + 1];
        const float xr1 = src[2 * p1], xi1 = src[2 * p1 + 1];
        const float r0 = xr0 * c0 - xi0 * s0, i0 = xr0 * s0 + xi0 * c0;
        const float r1 = xr1 * c1 - xi1 * s1, i1 = xr1 * s1 + xi1 * c1;
        float ss = r0 * r0 + i0 * i0 + r1 * r1 + i1 * i1;
        #pragma unroll
        for (int m = 16; m >= 1; m >>= 1) ss += __shfl_xor(ss, m, 32);
        const float inv = rsqrtf(ss * (1.0f / 128.0f) + 1e-6f);
        bf16* dst = q_t + (((size_t)(b * 32 + h)) * 2048 + l) * 128;
        dst[2 * p0] = (bf16)(r0 * inv); dst[2 * p0 + 1] = (bf16)(i0 * inv);
        dst[2 * p1] = (bf16)(r1 * inv); dst[2 * p1 + 1] = (bf16)(i1 * inv);
    }
    // ---- K: 8 heads, one per wave
    {
        const int h = wave;
        const float* src = k_raw + (size_t)bl * 1024 + h * 128;
        const float xr0 = src[2 * p0], xi0 = src[2 * p0 + 1];
        const float xr1 = src[2 * p1], xi1 = src[2 * p1 + 1];
        const float r0 = xr0 * c0 - xi0 * s0, i0 = xr0 * s0 + xi0 * c0;
        const float r1 = xr1 * c1 - xi1 * s1, i1 = xr1 * s1 + xi1 * c1;
        float ss = r0 * r0 + i0 * i0 + r1 * r1 + i1 * i1;
        #pragma unroll
        for (int m = 16; m >= 1; m >>= 1) ss += __shfl_xor(ss, m, 32);
        const float inv = rsqrtf(ss * (1.0f / 128.0f) + 1e-6f);
        bf16* dst = k_t + (((size_t)(b * 8 + h)) * 2048 + l) * 128;
        dst[2 * p0] = (bf16)(r0 * inv); dst[2 * p0 + 1] = (bf16)(i0 * inv);
        dst[2 * p1] = (bf16)(r1 * inv); dst[2 * p1 + 1] = (bf16)(i1 * inv);
    }
    // ---- V: straight convert + transpose (1024 elems / block)
    #pragma unroll
    for (int i = 0; i < 4; i++) {
        const int idx = tid + i * 256;
        const int h = idx >> 7, d = idx & 127;
        v_t[(((size_t)(b * 8 + h)) * 2048 + l) * 128 + d] =
            (bf16)v_raw[(size_t)bl * 1024 + idx];
    }
}

// ---------------------------------------------------------------------------
// Causal GQA flash attention. Grid (L/128, H, B), 8 waves/block.
// Wave owns 16 query rows; Q fragments resident in VGPRs; K/V blocks of 64
// keys staged in LDS (TDM tensor_load_to_lds when available, else uint4
// cooperative copy); S=QK^T and O+=PV via v_wmma_f32_16x16x32_bf16; online
// softmax with cross-lane reductions inside the 16-lane row groups.
// ---------------------------------------------------------------------------
__global__ void __launch_bounds__(256)
attn_fwd(const bf16* __restrict__ q_t, const bf16* __restrict__ k_t,
         const bf16* __restrict__ v_t, bf16* __restrict__ ctx)
{
    constexpr int LDK = 136;   // padded K/V LDS row stride (bf16) == TDM pad
    constexpr int LDP = 72;    // padded P LDS row stride (bf16)
    __shared__ bf16 Ks[64 * LDK];
    __shared__ bf16 Vs[64 * LDK];
    __shared__ bf16 Ps[8 * 16 * LDP];

    const int tid  = threadIdx.x;
    const int lane = tid & 31;
    const int wave = tid >> 5;
    const int lm   = lane & 15;
    const int hi   = lane >> 4;
    const int qt   = blockIdx.x;
    const int h    = blockIdx.y;
    const int b    = blockIdx.z;
    const int hkv  = h >> 2;                 // NREP = 4
    const int qbase = qt * 128 + wave * 16;

    const bf16* Qp = q_t + ((size_t)(b * 32 + h)) * 2048 * 128;
    const bf16* Kp = k_t + ((size_t)(b * 8 + hkv)) * 2048 * 128;
    const bf16* Vp = v_t + ((size_t)(b * 8 + hkv)) * 2048 * 128;

    // Q fragments for 4 k-chunks of D=128, resident for the whole kernel
    v16bf qf[4];
    #pragma unroll
    for (int kc = 0; kc < 4; kc++)
        #pragma unroll
        for (int v = 0; v < 8; v++) {
            const int kb = kc * 32 + ((v >> 2) << 4) + (hi << 3) + ((v & 3) << 1);
            qf[kc][2 * v]     = Qp[(size_t)(qbase + lm) * 128 + kb];
            qf[kc][2 * v + 1] = Qp[(size_t)(qbase + lm) * 128 + kb + 1];
        }

    float mrow[8], lrow[8];
    v8f o[8];
    #pragma unroll
    for (int j = 0; j < 8; j++) { mrow[j] = -1e30f; lrow[j] = 0.0f; }
    #pragma unroll
    for (int t = 0; t < 8; t++)
        #pragma unroll
        for (int e = 0; e < 8; e++) o[t][e] = 0.0f;

    const int nkb = (qt + 1) * 2;            // causal: key blocks of 64
    for (int kb = 0; kb < nkb; kb++) {
        // ---- stage K/V block (64 x 128 bf16 each) into LDS ----
#ifdef HAVE_TDM
        if (wave == 0) {
            tdm_load_2d_bf16((unsigned)(unsigned long long)(void*)Ks,
                             Kp + (size_t)kb * 64 * 128, 64);
            tdm_load_2d_bf16((unsigned)(unsigned long long)(void*)Vs,
                             Vp + (size_t)kb * 64 * 128, 64);
            __builtin_amdgcn_s_wait_tensorcnt(0);
        }
#else
        #pragma unroll
        for (int p = 0; p < 4; p++) {
            const int idx = tid + p * 256;   // 1024 x (8 bf16) chunks
            const int r = idx >> 4, c = idx & 15;
            *(uint4*)&Ks[r * LDK + c * 8] = *(const uint4*)(Kp + (size_t)(kb * 64 + r) * 128 + c * 8);
            *(uint4*)&Vs[r * LDK + c * 8] = *(const uint4*)(Vp + (size_t)(kb * 64 + r) * 128 + c * 8);
        }
#endif
        __syncthreads();

        // ---- S = Q * K^T  (16 q x 64 keys per wave) ----
        v8f s[4];
        #pragma unroll
        for (int nt = 0; nt < 4; nt++)
            #pragma unroll
            for (int e = 0; e < 8; e++) s[nt][e] = 0.0f;
        #pragma unroll
        for (int nt = 0; nt < 4; nt++)
            #pragma unroll
            for (int kc = 0; kc < 4; kc++) {
                v16bf bfr;
                #pragma unroll
                for (int v = 0; v < 8; v++) {
                    const int k = kc * 32 + ((v >> 2) << 4) + (hi << 3) + ((v & 3) << 1);
                    bfr[2 * v]     = Ks[(nt * 16 + lm) * LDK + k];
                    bfr[2 * v + 1] = Ks[(nt * 16 + lm) * LDK + k + 1];
                }
                s[nt] = WMMA_BF16(qf[kc], bfr, s[nt]);
            }

        // ---- scale, causal mask, online softmax (per row j) ----
        #pragma unroll
        for (int j = 0; j < 8; j++) {
            const int row = qbase + hi * 8 + j;
            float mx = -1e30f;
            #pragma unroll
            for (int nt = 0; nt < 4; nt++) {
                const int col = kb * 64 + nt * 16 + lm;
                float val = s[nt][j] * 0.08838834764831845f;
                if (col > row) val = -1e30f;
                s[nt][j] = val;
                mx = fmaxf(mx, val);
            }
            #pragma unroll
            for (int msk = 8; msk >= 1; msk >>= 1) mx = fmaxf(mx, __shfl_xor(mx, msk, 32));
            const float mnew = fmaxf(mrow[j], mx);
            const float corr = __expf(mrow[j] - mnew);
            float rs = 0.0f;
            #pragma unroll
            for (int nt = 0; nt < 4; nt++) {
                const float pv = __expf(s[nt][j] - mnew);
                s[nt][j] = pv;
                rs += pv;
            }
            #pragma unroll
            for (int msk = 8; msk >= 1; msk >>= 1) rs += __shfl_xor(rs, msk, 32);
            lrow[j] = lrow[j] * corr + rs;
            mrow[j] = mnew;
            #pragma unroll
            for (int t = 0; t < 8; t++) o[t][j] *= corr;
        }

        // ---- re-shape P: C/D layout -> A layout via wave-private LDS pad ----
        bf16* Pw = &Ps[wave * 16 * LDP];
        #pragma unroll
        for (int nt = 0; nt < 4; nt++)
            #pragma unroll
            for (int j = 0; j < 8; j++)
                Pw[(hi * 8 + j) * LDP + nt * 16 + lm] = (bf16)s[nt][j];

        // ---- O += P * V ----
        #pragma unroll
        for (int kc = 0; kc < 2; kc++) {
            v16bf pf;
            #pragma unroll
            for (int v = 0; v < 8; v++) {
                const int kbcol = kc * 32 + ((v >> 2) << 4) + (hi << 3) + ((v & 3) << 1);
                pf[2 * v]     = Pw[lm * LDP + kbcol];
                pf[2 * v + 1] = Pw[lm * LDP + kbcol + 1];
            }
            #pragma unroll
            for (int dt = 0; dt < 8; dt++) {
                v16bf vf;
                #pragma unroll
                for (int v = 0; v < 8; v++) {
                    const int k = kc * 32 + ((v >> 2) << 4) + (hi << 3) + ((v & 3) << 1);
                    vf[2 * v]     = Vs[k * LDK + dt * 16 + lm];
                    vf[2 * v + 1] = Vs[(k + 1) * LDK + dt * 16 + lm];
                }
                o[dt] = WMMA_BF16(pf, vf, o[dt]);
            }
        }
        __syncthreads();
    }

    // ---- normalize and write context, token-major bf16 for O-proj ----
    bf16* Cp = ctx + (size_t)b * 2048 * 4096 + (size_t)h * 128;
    #pragma unroll
    for (int dt = 0; dt < 8; dt++)
        #pragma unroll
        for (int j = 0; j < 8; j++) {
            const int row = qbase + hi * 8 + j;
            const float val = o[dt][j] / lrow[j];
            Cp[(size_t)row * 4096 + dt * 16 + lm] = (bf16)val;
        }
}

// ---------------------------------------------------------------------------
extern "C" void kernel_launch(void* const* d_in, const int* in_sizes, int n_in,
                              void* d_out, int out_size, void* d_ws, size_t ws_size,
                              hipStream_t stream)
{
    const float* hs   = (const float*)d_in[0];
    const float* wq   = (const float*)d_in[1];
    const float* wk   = (const float*)d_in[2];
    const float* wv   = (const float*)d_in[3];
    const float* wo   = (const float*)d_in[4];
    const float* fcos = (const float*)d_in[5];
    const float* fsin = (const float*)d_in[6];
    float* out = (float*)d_out;

    char* ws = (char*)d_ws;
    size_t off = 0;
    auto alloc = [&](size_t bytes) -> void* {
        void* p = ws + off;
        off += (bytes + 255) & ~(size_t)255;
        return p;
    };
    float* q_raw = (float*)alloc((size_t)16777216 * 4);   // B*L x H*D   f32
    float* k_raw = (float*)alloc((size_t)4194304  * 4);   // B*L x HKV*D f32
    float* v_raw = (float*)alloc((size_t)4194304  * 4);
    bf16*  q_t   = (bf16*) alloc((size_t)16777216 * 2);   // head-major bf16
    bf16*  k_t   = (bf16*) alloc((size_t)4194304  * 2);
    bf16*  v_t   = (bf16*) alloc((size_t)4194304  * 2);
    bf16*  ctx   = (bf16*) alloc((size_t)16777216 * 2);   // token-major bf16

    const dim3 blk(256);
    // QKV projections (M = B*L = 4096, K = HID = 4096)
    gemm_bf16_wmma<float><<<dim3(32, 32), blk, 0, stream>>>(hs, wq, q_raw, 4096, 4096, 4096);
    gemm_bf16_wmma<float><<<dim3(8, 32),  blk, 0, stream>>>(hs, wk, k_raw, 4096, 1024, 4096);
    gemm_bf16_wmma<float><<<dim3(8, 32),  blk, 0, stream>>>(hs, wv, v_raw, 4096, 1024, 4096);
    // RoPE + QK L2-norm + bf16 transpose
    rope_norm_transpose<<<dim3(4096), blk, 0, stream>>>(q_raw, k_raw, v_raw, fcos, fsin,
                                                        q_t, k_t, v_t);
    // Causal GQA flash attention (TDM-staged K/V)
    attn_fwd<<<dim3(16, 32, 2), blk, 0, stream>>>(q_t, k_t, v_t, ctx);
    // Output projection (bf16 context x f32 wo -> f32 out)
    gemm_bf16_wmma<bf16><<<dim3(32, 32), blk, 0, stream>>>(ctx, wo, out, 4096, 4096, 4096);
}
